// LSTMEncoder_6528350290265
// MI455X (gfx1250) — compile-verified, matrix operand order checked by reading, not run on hardware
//
#include <hip/hip_runtime.h>
#include <hip/hip_bf16.h>

// LSTM: B=32, S=4096, I=256, H=256, 4H=1024
#define B_ 32
#define S_ 4096
#define I_ 256
#define H_ 256
#define G_ 1024

typedef float  v8f   __attribute__((ext_vector_type(8)));
typedef __bf16 v16bf __attribute__((ext_vector_type(16)));
typedef __bf16 bf4   __attribute__((ext_vector_type(4)));
typedef float  f4    __attribute__((ext_vector_type(4)));
typedef unsigned int u4    __attribute__((ext_vector_type(4)));
typedef unsigned int u32x4 __attribute__((ext_vector_type(4)));
typedef int          i32x4 __attribute__((ext_vector_type(4)));
typedef int          i32x8 __attribute__((ext_vector_type(8)));

// ---- workspace layout (bytes) ----
#define WS_WIH   0u            // [1024][256] bf16 = 512KB
#define WS_WHH   524288u       // [1024][256] bf16 = 512KB
#define WS_BIAS  1048576u      // [1024] f32 (b_ih+b_hh) = 4KB
#define WS_H0    1052672u      // [32][256] bf16 = 16KB (double buffer 0)
#define WS_H1    1069056u      // [32][256] bf16 = 16KB (double buffer 1)
#define WS_CTR   1085440u      // barrier counter
#define WS_XBF   1114112u      // [32][4096][256] bf16 = 64MB (x pre-cast)

// ---- per-WG LDS layout (bytes), dynamic alloc ----
#define ROWS     264           // padded row stride in bf16 elems (512B data + 16B pad)
#define LDS_WIH  0             // 256 rows x 264 bf16 = 135168B (4 gates x 64 units of W_ih)
#define LDS_X0   135168        // x tile double-buffer 0: 32 x 264 bf16 = 16896B
#define LDS_X1   152064        // x tile double-buffer 1
#define LDS_HT   168960        // h_t tile: 32 x 264 bf16 = 16896B
#define LDS_TOT  185856        // ~181KB of the 320KB WGP LDS

#if __has_builtin(__builtin_amdgcn_tensor_load_to_lds)
#define TDM_OK 1
#endif

__device__ __forceinline__ v16bf ld_frag(const __bf16* p) {
  const u4* q = (const u4*)p;          // 16B-aligned
  union { u4 u[2]; v16bf v; } cv;
  cv.u[0] = q[0]; cv.u[1] = q[1];
  return cv.v;
}
__device__ __forceinline__ v8f splat8(float x) {
  v8f r = {x, x, x, x, x, x, x, x};
  return r;
}
__device__ __forceinline__ float sigmoidf_(float x) {
  return __builtin_amdgcn_rcpf(1.0f + __expf(-x));
}
__device__ __forceinline__ float tanhf_(float x) {
  float e = __expf(2.0f * x);
  return 1.0f - 2.0f * __builtin_amdgcn_rcpf(e + 1.0f);
}

#ifdef TDM_OK
// 2D bf16 tile DMA: nRows rows of rowElems bf16, global row stride rowStrideElems,
// into LDS at lds_off with 16B padding appended to every 512B of data (ROWS layout).
__device__ __forceinline__ void tdm_load_tile_bf16(const void* gptr, unsigned lds_off,
                                                   unsigned rowElems, unsigned nRows,
                                                   unsigned long long rowStrideElems) {
  unsigned long long ga = (unsigned long long)(uintptr_t)gptr;
  u32x4 g0;
  g0.x = 1u;                                            // count=1, no gather
  g0.y = lds_off;                                       // lds_addr
  g0.z = (unsigned)(ga & 0xFFFFFFFFu);                  // global_addr[31:0]
  g0.w = (unsigned)((ga >> 32) & 0x1FFFFFFu) | (2u << 30);  // global_addr[56:32] | type=2
  i32x8 g1;
  g1.s0 = (int)((1u << 16)        // data_size = 2B
              | (1u << 20)        // pad_enable
              | (6u << 22)        // pad_interval: every 128 DWORDs (=512B)
              | (3u << 25));      // pad_amount: 4 DWORDs (=16B)
  g1.s1 = (int)(rowElems << 16);  // tensor_dim0[15:0] (abar addr = 0)
  g1.s2 = (int)(nRows << 16);     // tensor_dim0[31:16]=0, tensor_dim1[15:0]
  g1.s3 = (int)(rowElems << 16);  // tensor_dim1[31:16]=0, tile_dim0
  g1.s4 = (int)nRows;             // tile_dim1, tile_dim2=0
  g1.s5 = (int)(unsigned)(rowStrideElems & 0xFFFFFFFFull);        // dim0_stride[31:0]
  g1.s6 = (int)(unsigned)((rowStrideElems >> 32) & 0xFFFFull);    // dim0_stride[47:32]
  g1.s7 = 0;
  i32x4 gz = {0, 0, 0, 0};
#if __clang_major__ >= 23
  i32x8 gz8 = {0, 0, 0, 0, 0, 0, 0, 0};
  __builtin_amdgcn_tensor_load_to_lds(g0, g1, gz, gz, gz8, 0);
#else
  __builtin_amdgcn_tensor_load_to_lds(g0, g1, gz, gz, 0);
#endif
}
#endif

// -------- prep: bf16-cast weights, fuse biases, seed h0, reset barrier --------
__global__ void lstm_prep(const float* __restrict__ Wih, const float* __restrict__ Whh,
                          const float* __restrict__ bih, const float* __restrict__ bhh,
                          const float* __restrict__ h0, char* __restrict__ ws) {
  int t = blockIdx.x * 256 + threadIdx.x;
  __bf16* wih_b = (__bf16*)(ws + WS_WIH);
  __bf16* whh_b = (__bf16*)(ws + WS_WHH);
  float*  bias  = (float*)(ws + WS_BIAS);
  __bf16* h0b   = (__bf16*)(ws + WS_H0);
  unsigned* ctr = (unsigned*)(ws + WS_CTR);
  if (t < G_ * I_) { wih_b[t] = (__bf16)Wih[t]; whh_b[t] = (__bf16)Whh[t]; }
  if (t < G_)      bias[t] = bih[t] + bhh[t];
  if (t < B_ * H_) h0b[t] = (__bf16)h0[t];
  if (t == 0)      *ctr = 0u;
}

// -------- x pre-cast: f32 -> bf16, 4 elems/thread --------
__global__ void lstm_xcast(const float* __restrict__ x, char* __restrict__ ws) {
  size_t t = (size_t)blockIdx.x * 256 + threadIdx.x;
  __bf16* xb = (__bf16*)(ws + WS_XBF);
  f4 v = *(const f4*)(x + t * 4);
  bf4 bv;
  bv.x = (__bf16)v.x; bv.y = (__bf16)v.y; bv.z = (__bf16)v.z; bv.w = (__bf16)v.w;
  *(bf4*)(xb + t * 4) = bv;
}

// -------- persistent scan: 4 WGs x 256 threads; WG wg owns hidden units [wg*64, wg*64+64) --------
__global__ void __launch_bounds__(256, 1) lstm_scan(const float* __restrict__ c0,
                                                    char* __restrict__ ws,
                                                    float* __restrict__ out) {
  extern __shared__ char smem[];
  __bf16* ldsWih = (__bf16*)(smem + LDS_WIH);
  __bf16* ldsH   = (__bf16*)(smem + LDS_HT);

  const int tid   = threadIdx.x;
  const int wg    = blockIdx.x;      // 0..3
  const int lane  = tid & 31;
  const int wave  = tid >> 5;        // 0..7
  const int mtile = wave >> 2;       // 0..1  (batch rows 16*mtile..)
  const int jtile = wave & 3;        // 0..3  (16 hidden units within this WG's 64)
  const int l15   = lane & 15;
  const int khalf = lane >> 4;
  const int jcol  = wg * 64 + jtile * 16 + l15;   // global hidden unit
  const int arow  = mtile * 16 + l15;             // A-fragment row (batch)
  const int wrow  = jtile * 16 + l15;             // B-fragment row within a gate block

  const __bf16* gWih  = (const __bf16*)(ws + WS_WIH);
  const __bf16* gWhh  = (const __bf16*)(ws + WS_WHH);
  const __bf16* gXbf  = (const __bf16*)(ws + WS_XBF);
  const float*  gBias = (const float*)(ws + WS_BIAS);
  __bf16* hbuf0 = (__bf16*)(ws + WS_H0);
  __bf16* hbuf1 = (__bf16*)(ws + WS_H1);
  unsigned* ctr = (unsigned*)(ws + WS_CTR);

  // Stage this WG's W_ih slice (gate q, units wg*64..+63) into LDS, padded rows.
  for (int idx = tid; idx < 256 * 32; idx += 256) {
    int r = idx >> 5, ch = idx & 31;            // LDS row r = q*64+jj, 16B chunk ch
    int q = r >> 6, jj = r & 63;
    int g = q * 256 + wg * 64 + jj;             // global gate row
    *(u4*)(ldsWih + r * ROWS + ch * 8) = *(const u4*)(gWih + g * 256 + ch * 8);
  }

#ifdef TDM_OK
  // Kick off the DMA of x tile for step 0 into buffer 0 (overlaps with setup).
  if (wave == 0)
    tdm_load_tile_bf16(gXbf, LDS_X0, I_, B_, (unsigned long long)S_ * I_);
#endif

  // W_hh fragments are loop-invariant: cache all 32 (8 kt x 4 gates) in VGPRs,
  // loaded straight from global (once). 32 x v16bf = 256 VGPRs/lane.
  v16bf whhF[32];
#pragma unroll
  for (int kt = 0; kt < 8; ++kt) {
#pragma unroll
    for (int q = 0; q < 4; ++q) {
      int g = q * 256 + wg * 64 + wrow;
      whhF[kt * 4 + q] = ld_frag(gWhh + g * 256 + kt * 32 + khalf * 16);
    }
  }

  // Per-lane loop constants: fused biases for this column, persistent cell tile from c0.
  const float b_i = gBias[0 * 256 + jcol];
  const float b_f = gBias[1 * 256 + jcol];
  const float b_g = gBias[2 * 256 + jcol];
  const float b_o = gBias[3 * 256 + jcol];
  v8f creg;
#pragma unroll
  for (int r = 0; r < 8; ++r) {
    int m = mtile * 16 + (khalf << 3) + r;
    creg[r] = c0[m * H_ + jcol];
  }

  __syncthreads();

  for (int s = 0; s < S_; ++s) {
    const __bf16* hsrc = (s & 1) ? hbuf1 : hbuf0;
    const unsigned xcurOff = (s & 1) ? LDS_X1 : LDS_X0;   // this step's x buffer
    const unsigned xnxtOff = (s & 1) ? LDS_X0 : LDS_X1;   // next step's x buffer
#ifdef TDM_OK
    // Wave 0: keep the x pipeline one step ahead (double-buffered LDS tile);
    // its TDM ops complete in order, so tensorcnt<=1 means x(s) has landed.
    // Wave 1: DMA this step's h tile (true sequential dependency).
    if (wave == 0) {
      if (s + 1 < S_) {
        tdm_load_tile_bf16(gXbf + (size_t)(s + 1) * I_, xnxtOff,
                           I_, B_, (unsigned long long)S_ * I_);
        __builtin_amdgcn_s_wait_tensorcnt(1);
      } else {
        __builtin_amdgcn_s_wait_tensorcnt(0);
      }
    } else if (wave == 1) {
      tdm_load_tile_bf16(hsrc, LDS_HT, H_, B_, (unsigned long long)H_);
      __builtin_amdgcn_s_wait_tensorcnt(0);
    }
#else
    {
      __bf16* xdst = (__bf16*)(smem + xcurOff);
      for (int idx = tid; idx < 32 * 32; idx += 256) {
        int b = idx >> 5, ch = idx & 31;
        *(u4*)(xdst + b * ROWS + ch * 8) =
            *(const u4*)(gXbf + ((size_t)b * S_ + s) * I_ + ch * 8);
        *(u4*)(ldsH + b * ROWS + ch * 8) = *(const u4*)(hsrc + b * 256 + ch * 8);
      }
    }
#endif
    // Warm L2 two steps ahead (one step ahead is already in flight via TDM).
    if (tid < 32 && s + 2 < S_)
      __builtin_prefetch(gXbf + ((size_t)tid * S_ + (s + 2)) * I_, 0, 3);
    __syncthreads();

    const __bf16* xcur = (const __bf16*)(smem + xcurOff);

    // gates = bias + x_t @ W_ih^T + h_t @ W_hh^T ; software-pipelined over kt.
    v8f acc_i = splat8(b_i), acc_f = splat8(b_f), acc_g = splat8(b_g), acc_o = splat8(b_o);
    v16bf aX = ld_frag(xcur + arow * ROWS + khalf * 16);
    v16bf aH = ld_frag(ldsH + arow * ROWS + khalf * 16);
    v16bf wi[4];
#pragma unroll
    for (int q = 0; q < 4; ++q)
      wi[q] = ld_frag(ldsWih + (q * 64 + wrow) * ROWS + khalf * 16);

#pragma unroll
    for (int kt = 0; kt < 8; ++kt) {
      v16bf aXn, aHn, win[4];
      if (kt < 7) {                      // issue next-kt loads before this kt's WMMAs
        const int ko = (kt + 1) * 32 + khalf * 16;
        aXn = ld_frag(xcur + arow * ROWS + ko);
        aHn = ld_frag(ldsH + arow * ROWS + ko);
#pragma unroll
        for (int q = 0; q < 4; ++q)
          win[q] = ld_frag(ldsWih + (q * 64 + wrow) * ROWS + ko);
      }
      acc_i = __builtin_amdgcn_wmma_f32_16x16x32_bf16(false, aX, false, wi[0], (short)0, acc_i, false, false);
      acc_f = __builtin_amdgcn_wmma_f32_16x16x32_bf16(false, aX, false, wi[1], (short)0, acc_f, false, false);
      acc_g = __builtin_amdgcn_wmma_f32_16x16x32_bf16(false, aX, false, wi[2], (short)0, acc_g, false, false);
      acc_o = __builtin_amdgcn_wmma_f32_16x16x32_bf16(false, aX, false, wi[3], (short)0, acc_o, false, false);
      acc_i = __builtin_amdgcn_wmma_f32_16x16x32_bf16(false, aH, false, whhF[kt * 4 + 0], (short)0, acc_i, false, false);
      acc_f = __builtin_amdgcn_wmma_f32_16x16x32_bf16(false, aH, false, whhF[kt * 4 + 1], (short)0, acc_f, false, false);
      acc_g = __builtin_amdgcn_wmma_f32_16x16x32_bf16(false, aH, false, whhF[kt * 4 + 2], (short)0, acc_g, false, false);
      acc_o = __builtin_amdgcn_wmma_f32_16x16x32_bf16(false, aH, false, whhF[kt * 4 + 3], (short)0, acc_o, false, false);
      aX = aXn; aH = aHn;
#pragma unroll
      for (int q = 0; q < 4; ++q) wi[q] = win[q];
    }

    // Elementwise LSTM cell update; c stays in registers across all 4096 steps.
    __bf16* hdst = (s & 1) ? hbuf0 : hbuf1;
#pragma unroll
    for (int r = 0; r < 8; ++r) {
      float ig = sigmoidf_(acc_i[r]);
      float fg = sigmoidf_(acc_f[r]);
      float gg = tanhf_(acc_g[r]);
      float og = sigmoidf_(acc_o[r]);
      float c  = fg * creg[r] + ig * gg;
      creg[r]  = c;
      float h  = og * tanhf_(c);
      int m = mtile * 16 + (khalf << 3) + r;     // batch row
      out[((size_t)m * S_ + s) * H_ + jcol] = h; // f32 result
      hdst[m * 256 + jcol] = (__bf16)h;          // bf16 h for next step
    }

    // Make h visible device-wide, then 4-block barrier (monotone counter).
    __threadfence();
    __syncthreads();
    if (tid == 0) {
      __hip_atomic_fetch_add(ctr, 1u, __ATOMIC_RELEASE, __HIP_MEMORY_SCOPE_AGENT);
      unsigned tgt = 4u * (unsigned)(s + 1);
      while (__hip_atomic_load(ctr, __ATOMIC_ACQUIRE, __HIP_MEMORY_SCOPE_AGENT) < tgt)
        __builtin_amdgcn_s_sleep(1);
    }
    __builtin_amdgcn_s_cluster_barrier();  // NOP if not cluster-launched; real sync if it is
    __syncthreads();
  }
}

extern "C" void kernel_launch(void* const* d_in, const int* in_sizes, int n_in,
                              void* d_out, int out_size, void* d_ws, size_t ws_size,
                              hipStream_t stream) {
  const float* x   = (const float*)d_in[0];
  const float* h0  = (const float*)d_in[1];
  const float* c0  = (const float*)d_in[2];
  const float* Wih = (const float*)d_in[3];
  const float* Whh = (const float*)d_in[4];
  const float* bih = (const float*)d_in[5];
  const float* bhh = (const float*)d_in[6];
  float* out = (float*)d_out;
  char*  ws  = (char*)d_ws;

  lstm_prep<<<1024, 256, 0, stream>>>(Wih, Whh, bih, bhh, h0, ws);
  lstm_xcast<<<(B_ * S_ * I_ / 4 + 255) / 256, 256, 0, stream>>>(x, ws);
  lstm_scan<<<4, 256, LDS_TOT, stream>>>(c0, ws, out);
}
